// NVGPTAttention_49486613184800
// MI455X (gfx1250) — matrix operand order repaired and checked by reading, not compiled, not run on hardware
//
#include <hip/hip_runtime.h>

typedef __bf16 bf16_t;
typedef __attribute__((ext_vector_type(16))) __bf16 v16bf;
typedef __attribute__((ext_vector_type(8)))  __bf16 v8bf;
typedef __attribute__((ext_vector_type(8)))  float  v8f;

#define SEQ    2048
#define BATCH  2
#define HIDDEN 2048
#define NHEADS 16
#define HDIM   128
#define ROWS   (SEQ * BATCH)   // 4096
#define QKV_N  (3 * HIDDEN)    // 6144

// ---------- helpers ----------

__device__ __forceinline__ bf16_t f2bf(float f) {
  union { float f; unsigned u; } a; a.f = f;
  unsigned r = (a.u + 0x7FFFu + ((a.u >> 16) & 1u)) >> 16;   // RNE
  union { unsigned short s; bf16_t b; } c; c.s = (unsigned short)r;
  return c.b;
}

// A-matrix 16x32 bf16 fragment (ISA layout): lane L holds row M=L%16,
// K chunks [8h,8h+8) and [16+8h,16+8h+8) with h=L/16. base pre-offset to row 0.
__device__ __forceinline__ v16bf load_a_frag(const bf16_t* base, int ld, int k0, int lane) {
  int m = lane & 15;
  int h = lane >> 4;
  const bf16_t* p = base + (size_t)m * ld + (k0 + 8 * h);
  v8bf lo = *(const v8bf*)p;
  v8bf hi = *(const v8bf*)(p + 16);
  v16bf r;
#pragma unroll
  for (int i = 0; i < 8; ++i) { r[i] = lo[i]; r[i + 8] = hi[i]; }
  return r;
}

// B-matrix 32x16 bf16 fragment: lane L holds column N=L%16,
// K chunks [8h,8h+8) and [16+8h,16+8h+8). Source row n (contraction-contiguous), base pre-offset to n0.
__device__ __forceinline__ v16bf load_b_frag(const bf16_t* base, int ld, int k0, int lane) {
  int n = lane & 15;
  int h = lane >> 4;
  const bf16_t* p = base + (size_t)n * ld + (k0 + 8 * h);
  v8bf lo = *(const v8bf*)p;
  v8bf hi = *(const v8bf*)(p + 16);
  v16bf r;
#pragma unroll
  for (int i = 0; i < 8; ++i) { r[i] = lo[i]; r[i + 8] = hi[i]; }
  return r;
}

__device__ __forceinline__ v8f wmma_bf16(v16bf a, v16bf b, v8f c) {
  return __builtin_amdgcn_wmma_f32_16x16x32_bf16(false, a, false, b, (short)0, c, false, false);
}

// Async-copy a ROWS_T x 32 bf16 tile (row-major, leading dim ldg) from global
// into LDS at byte offset lds_base. ROWS_T*4 x 16B chunks spread over 256 threads.
// Tracked by ASYNCcnt; caller must s_wait_asynccnt + barrier.
template <int ROWS_T>
__device__ __forceinline__ void async_tile(const bf16_t* gbase, int ldg,
                                           unsigned lds_base, int tid) {
#pragma unroll
  for (int r = 0; r < ROWS_T / 64; ++r) {
    int i = tid + 256 * r;
    int row = i >> 2;
    int ch  = i & 3;
    unsigned long long ga =
        (unsigned long long)(const void*)(gbase + (size_t)row * ldg + ch * 8);
    unsigned lds = lds_base + (unsigned)(row * 64 + ch * 16);
    asm volatile("global_load_async_to_lds_b128 %0, %1, off"
                 :: "v"(lds), "v"(ga) : "memory");
  }
}

// ---------- kernels ----------

__global__ void cvt_f32_bf16(const float* __restrict__ in, bf16_t* __restrict__ out, int n) {
  int i = blockIdx.x * blockDim.x + threadIdx.x;
  int stride = gridDim.x * blockDim.x;
  for (; i < n; i += stride) out[i] = f2bf(in[i]);
}

// LDS-staged, double-buffered WMMA GEMM. Block tile 128(M) x 256(N),
// 8 waves as 2(M) x 4(N), wave tile 64x64 => 16 WMMAs per 16 ds_load_b128.
// C-layout (16x16 f32 C/D): VGPR j, lane L -> M = j + 8*(L/16), N = L%16.
__global__ __launch_bounds__(256) void gemm_wmma_bf16(
    const bf16_t* __restrict__ A, const bf16_t* __restrict__ W,
    const float* __restrict__ bias, float* __restrict__ outF,
    bf16_t* __restrict__ qb, bf16_t* __restrict__ kb, bf16_t* __restrict__ vtb,
    int N, int K, int mode) {
  __shared__ __align__(16) bf16_t tA[2][128 * 32];   // 2 x 8KB
  __shared__ __align__(16) bf16_t tB[2][256 * 32];   // 2 x 16KB

  const int tid   = threadIdx.x;
  const int lane  = tid & 31;
  const int wave  = tid >> 5;
  const int waveM = wave & 1;
  const int waveN = wave >> 1;
  const int m0l = waveM * 64;   // wave tile inside block
  const int n0l = waveN * 64;
  const int m0 = blockIdx.y * 128 + m0l;
  const int n0 = blockIdx.x * 256 + n0l;

  const bf16_t* Ab = A + (size_t)(blockIdx.y * 128) * K;
  const bf16_t* Wb = W + (size_t)(blockIdx.x * 256) * K;

  v8f acc[4][4] = {};

  // prologue: stage first K-slice into buffer 0
  async_tile<128>(Ab, K, (unsigned)(unsigned long long)(void*)&tA[0][0], tid);
  async_tile<256>(Wb, K, (unsigned)(unsigned long long)(void*)&tB[0][0], tid);

  for (int k0 = 0; k0 < K; k0 += 32) {
    const int buf = (k0 >> 5) & 1;
    asm volatile("s_wait_asynccnt 0x0" ::: "memory");
    __syncthreads();

    if (k0 + 32 < K) {
      const int nb = buf ^ 1;
      async_tile<128>(Ab + (k0 + 32), K,
                      (unsigned)(unsigned long long)(void*)&tA[nb][0], tid);
      async_tile<256>(Wb + (k0 + 32), K,
                      (unsigned)(unsigned long long)(void*)&tB[nb][0], tid);
    }

    // load all fragments first (distinct regs -> staggered dscnt waits)
    v16bf af[4], bfr[4];
#pragma unroll
    for (int mt = 0; mt < 4; ++mt)
      af[mt] = load_a_frag(&tA[buf][(size_t)(m0l + 16 * mt) * 32], 32, 0, lane);
#pragma unroll
    for (int t = 0; t < 4; ++t)
      bfr[t] = load_b_frag(&tB[buf][(size_t)(n0l + 16 * t) * 32], 32, 0, lane);
#pragma unroll
    for (int mt = 0; mt < 4; ++mt)
#pragma unroll
      for (int t = 0; t < 4; ++t)
        acc[mt][t] = wmma_bf16(af[mt], bfr[t], acc[mt][t]);
  }

  const int mloc  = 8 * (lane >> 4);
  const int nlane = lane & 15;
#pragma unroll
  for (int mt = 0; mt < 4; ++mt) {
#pragma unroll
    for (int t = 0; t < 4; ++t) {
      int n = n0 + 16 * t + nlane;
      float bv = bias[n];
#pragma unroll
      for (int j = 0; j < 8; ++j) {
        int m = m0 + 16 * mt + mloc + j;
        float val = acc[mt][t][j] + bv;
        if (mode == 0) {
          outF[(size_t)m * N + n] = val;
        } else {
          // mixed[r][n], n -> (head, {q,k,v}, d): h = n/384, typ = (n%384)/128
          int s = m >> 1, b = m & 1;              // r = s*BATCH + b, BATCH==2
          int hh = n / 384;
          int inner = n - hh * 384;
          int typ = inner >> 7;
          int dc  = inner & 127;
          bf16_t bw = f2bf(val);
          size_t bh = (size_t)(b * NHEADS + hh);
          if (typ == 0)      qb [(bh * SEQ + s) * HDIM + dc] = bw;      // Q [b,h,s,d]
          else if (typ == 1) kb [(bh * SEQ + s) * HDIM + dc] = bw;      // K [b,h,s,d]
          else               vtb[(bh * HDIM + dc) * SEQ + s] = bw;      // V^T [b,h,d,s]
        }
      }
    }
  }
}

__global__ __launch_bounds__(256) void flash_attn_bf16(
    const bf16_t* __restrict__ qb, const bf16_t* __restrict__ kb,
    const bf16_t* __restrict__ vtb, bf16_t* __restrict__ ctx) {
  __shared__ __align__(16) bf16_t psh[8 * 16 * 64];   // 2KB per wave

  const int lane = threadIdx.x & 31;
  const int wave = threadIdx.x >> 5;
  bf16_t* pp = psh + wave * (16 * 64);

  const int task  = blockIdx.x * 8 + wave;   // 0..4095
  const int qtile = task & 127;
  const int comb  = task >> 7;               // b*NHEADS + h
  const int bb = comb >> 4;
  const int hh = comb & 15;
  const int qs = qtile * 16;

  const bf16_t* qbase = qb  + ((size_t)comb * SEQ + qs) * HDIM;
  const bf16_t* kbase = kb  + (size_t)comb * SEQ * HDIM;
  const bf16_t* vbase = vtb + (size_t)comb * HDIM * SEQ;

  v16bf qf[4];
#pragma unroll
  for (int kc = 0; kc < 4; ++kc) qf[kc] = load_a_frag(qbase, HDIM, kc * 32, lane);

  v8f o[8] = {};
  float mrow[8], lrow[8];
#pragma unroll
  for (int j = 0; j < 8; ++j) { mrow[j] = -1e30f; lrow[j] = 0.f; }

  const float scale = 0.08838834764831845f;  // 1/sqrt(HDIM)
  const int mloc  = 8 * (lane >> 4);
  const int nlane = lane & 15;

  for (int kt = 0; kt < qs + 16; kt += 64) {
    if (kt + 64 < qs + 16)
      __builtin_prefetch(kbase + (size_t)(kt + 64) * HDIM + lane * 16, 0, 1);

    // S = Q K^T for 16 q rows x 64 keys
    v8f sacc[4] = {};
#pragma unroll
    for (int t = 0; t < 4; ++t) {
      const bf16_t* kt_p = kbase + (size_t)(kt + 16 * t) * HDIM;
#pragma unroll
      for (int kc = 0; kc < 4; ++kc) {
        v16bf kf = load_b_frag(kt_p, HDIM, kc * 32, lane);
        sacc[t] = wmma_bf16(qf[kc], kf, sacc[t]);
      }
    }
    // scale + causal mask
#pragma unroll
    for (int t = 0; t < 4; ++t) {
      int key = kt + 16 * t + nlane;
#pragma unroll
      for (int j = 0; j < 8; ++j) {
        float sv = sacc[t][j] * scale;
        int qrow = qs + mloc + j;
        sacc[t][j] = (key > qrow) ? -1e30f : sv;
      }
    }
    // online softmax: row reductions across the 16-lane half (rows differ per half)
    float rmax[8];
#pragma unroll
    for (int j = 0; j < 8; ++j)
      rmax[j] = fmaxf(fmaxf(sacc[0][j], sacc[1][j]), fmaxf(sacc[2][j], sacc[3][j]));
#pragma unroll
    for (int off = 1; off < 16; off <<= 1)
#pragma unroll
      for (int j = 0; j < 8; ++j)
        rmax[j] = fmaxf(rmax[j], __shfl_xor(rmax[j], off, 32));

    float corr[8];
#pragma unroll
    for (int j = 0; j < 8; ++j) {
      float mn = fmaxf(mrow[j], rmax[j]);
      corr[j] = __expf(mrow[j] - mn);
      mrow[j] = mn;
    }
    float rsum[8];
#pragma unroll
    for (int j = 0; j < 8; ++j) rsum[j] = 0.f;
#pragma unroll
    for (int t = 0; t < 4; ++t)
#pragma unroll
      for (int j = 0; j < 8; ++j) {
        float pv = __expf(sacc[t][j] - mrow[j]);
        sacc[t][j] = pv;
        rsum[j] += pv;
      }
#pragma unroll
    for (int off = 1; off < 16; off <<= 1)
#pragma unroll
      for (int j = 0; j < 8; ++j)
        rsum[j] += __shfl_xor(rsum[j], off, 32);
#pragma unroll
    for (int j = 0; j < 8; ++j) lrow[j] = lrow[j] * corr[j] + rsum[j];
#pragma unroll
    for (int n = 0; n < 8; ++n)
#pragma unroll
      for (int j = 0; j < 8; ++j) o[n][j] *= corr[j];

    // P through LDS: C-layout -> A-layout (bf16)
#pragma unroll
    for (int t = 0; t < 4; ++t)
#pragma unroll
      for (int j = 0; j < 8; ++j)
        pp[(mloc + j) * 64 + 16 * t + nlane] = f2bf(sacc[t][j]);

    asm volatile("s_wait_dscnt 0" ::: "memory");

    v16bf pf0 = load_a_frag(pp, 64, 0, lane);
    v16bf pf1 = load_a_frag(pp, 64, 32, lane);

    // O += P V  (B-frags from V^T rows = contiguous keys)
#pragma unroll
    for (int n = 0; n < 8; ++n) {
      const bf16_t* vb_n = vbase + (size_t)(16 * n) * SEQ;
      v16bf vf0 = load_b_frag(vb_n, SEQ, kt, lane);
      v16bf vf1 = load_b_frag(vb_n, SEQ, kt + 32, lane);
      o[n] = wmma_bf16(pf0, vf0, o[n]);
      o[n] = wmma_bf16(pf1, vf1, o[n]);
    }
    asm volatile("s_wait_dscnt 0" ::: "memory");  // P reads done before next-iter stores
  }

  // normalize, write ctx [s,b,H] bf16 (A of the output projection)
#pragma unroll
  for (int j = 0; j < 8; ++j) {
    int s = qs + mloc + j;
    float inv = 1.0f / lrow[j];
    size_t rowoff = ((size_t)s * BATCH + bb) * HIDDEN + (size_t)hh * HDIM;
#pragma unroll
    for (int n = 0; n < 8; ++n)
      ctx[rowoff + 16 * n + nlane] = f2bf(o[n][j] * inv);
  }
}

// ---------- launch ----------

extern "C" void kernel_launch(void* const* d_in, const int* in_sizes, int n_in,
                              void* d_out, int out_size, void* d_ws, size_t ws_size,
                              hipStream_t stream) {
  (void)in_sizes; (void)n_in; (void)out_size; (void)ws_size;
  const float* hs      = (const float*)d_in[0];  // [2048,2,2048]
  const float* w_qkv   = (const float*)d_in[1];  // [6144,2048]
  const float* b_qkv   = (const float*)d_in[2];  // [6144]
  const float* w_dense = (const float*)d_in[3];  // [2048,2048]
  const float* b_dense = (const float*)d_in[4];  // [2048]
  float* out = (float*)d_out;

  char* p = (char*)d_ws;
  bf16_t* hsb   = (bf16_t*)p; p += (size_t)ROWS * HIDDEN * 2;
  bf16_t* wqb   = (bf16_t*)p; p += (size_t)QKV_N * HIDDEN * 2;
  bf16_t* wdb   = (bf16_t*)p; p += (size_t)HIDDEN * HIDDEN * 2;
  bf16_t* qbuf  = (bf16_t*)p; p += (size_t)ROWS * HIDDEN * 2;
  bf16_t* kbuf  = (bf16_t*)p; p += (size_t)ROWS * HIDDEN * 2;
  bf16_t* vtbuf = (bf16_t*)p; p += (size_t)ROWS * HIDDEN * 2;
  bf16_t* ctxb  = (bf16_t*)p; p += (size_t)ROWS * HIDDEN * 2;

  cvt_f32_bf16<<<1024, 256, 0, stream>>>(hs, hsb, ROWS * HIDDEN);
  cvt_f32_bf16<<<1024, 256, 0, stream>>>(w_qkv, wqb, QKV_N * HIDDEN);
  cvt_f32_bf16<<<1024, 256, 0, stream>>>(w_dense, wdb, HIDDEN * HIDDEN);

  dim3 g1(QKV_N / 256, ROWS / 128);
  gemm_wmma_bf16<<<g1, 256, 0, stream>>>(hsb, wqb, b_qkv, nullptr,
                                         qbuf, kbuf, vtbuf, QKV_N, HIDDEN, 1);

  flash_attn_bf16<<<512, 256, 0, stream>>>(qbuf, kbuf, vtbuf, ctxb);

  dim3 g2(HIDDEN / 256, ROWS / 128);
  gemm_wmma_bf16<<<g2, 256, 0, stream>>>(ctxb, wdb, b_dense, out,
                                         nullptr, nullptr, nullptr, HIDDEN, HIDDEN, 0);
}
